// GraphMambaLastGate_82652350644777
// MI455X (gfx1250) — matrix operand order repaired and checked by reading, not compiled, not run on hardware
//
#include <hip/hip_runtime.h>
#include <hip/hip_bf16.h>
#include <math.h>

// ---------------------------------------------------------------------------
// Problem constants (from reference)
// ---------------------------------------------------------------------------
#define NN   1024          // nodes
#define TT   32            // timesteps
#define HH   256           // hidden
#define PDD  256           // pos dim
#define EE   512           // mamba expand (2*H)
#define DTR  16            // dt_rank
#define SDD  16            // state dim
#define ROWS (NN * TT)     // 32768 rows for [T,N,*] or [N,T,*] flattened

typedef _Float16 half_t;
typedef __attribute__((ext_vector_type(16))) half_t v16h;
typedef __attribute__((ext_vector_type(8)))  float  v8f;

struct __align__(16) U4 { unsigned int a, b, c, d; };
union V16 { v16h v; U4 u[2]; };

// Epilogue modes
#define EPI_PLAIN  0
#define EPI_RELU   1
#define EPI_F16    2
#define EPI_RES    3
#define EPI_SCORES 4

// low 32 bits of a generic shared-aperture address == wave-relative LDS byte addr
__device__ __forceinline__ unsigned lds_addr32(const void* p)
{
    return (unsigned)(size_t)p;
}

// ---------------------------------------------------------------------------
// Generic tiled WMMA GEMM:  C[M,N] = A[M,K] * B^T[N,K]^T  (+ epilogue)
// A, BT are f16 row-major; lda/ldb are row strides (elements); batched via z.
// Block tile 128x64, 256 threads = 8 waves; wave tile 32x32 = 2x2 WMMA tiles.
// K must be a multiple of 32; lda/ldb must be multiples of 8.
// Staging: double-buffered GLOBAL_LOAD_ASYNC_TO_LDS_B128 (ASYNCcnt-tracked),
// next K-slab streams into LDS while current slab feeds the WMMAs.
// ---------------------------------------------------------------------------
template <int EPI>
__global__ __launch_bounds__(256)
void wmma_gemm_kernel(const half_t* __restrict__ A, long lda, long sA,
                      const half_t* __restrict__ B, long ldb, long sB,
                      float* __restrict__ C, long ldc, long sC,
                      half_t* __restrict__ C16,
                      const float* __restrict__ bias,
                      const float* __restrict__ resid, long ldr, long sR,
                      const float* __restrict__ shift_ptr, float scale,
                      float* __restrict__ C2,
                      int M, int Nn, int K)
{
    __shared__ half_t As[2][128 * 32];   // 2 x 8 KB
    __shared__ half_t Bs[2][64 * 32];    // 2 x 4 KB

    const int tid  = threadIdx.x;
    const int lane = tid & 31;
    const int w    = tid >> 5;
    const int wm   = w & 3;           // wave row 0..3   (32 rows each)
    const int wn   = w >> 2;          // wave col 0..1   (32 cols each)
    const int bm   = blockIdx.y * 128;
    const int bn   = blockIdx.x * 64;
    const long z   = blockIdx.z;

    A += z * sA;
    B += z * sB;

    // staging assignment: A tile = 512 chunks of 8 halfs, 2 per thread;
    // B tile = 256 chunks, 1 per thread.
    const int arow0 = tid >> 2;                 // 0..63
    const int arow1 = arow0 + 64;               // 64..127
    const int acol  = (tid & 3) << 3;           // 0,8,16,24
    const int brow  = tid >> 2;                 // 0..63
    const int bcol  = acol;

    const bool a0_in = (bm + arow0) < M;
    const bool a1_in = (bm + arow1) < M;
    const bool b_in  = (bn + brow) < Nn;

    const half_t* gA0 = A + (long)(bm + arow0) * lda + acol;
    const half_t* gA1 = A + (long)(bm + arow1) * lda + acol;
    const half_t* gB  = B + (long)(bn + brow) * ldb + bcol;

    unsigned ldsA0[2], ldsA1[2], ldsB[2];
    #pragma unroll
    for (int b = 0; b < 2; ++b) {
        ldsA0[b] = lds_addr32(&As[b][arow0 * 32 + acol]);
        ldsA1[b] = lds_addr32(&As[b][arow1 * 32 + acol]);
        ldsB[b]  = lds_addr32(&Bs[b][brow * 32 + bcol]);
    }

    // pre-zero LDS slots owned by out-of-bounds lanes (stay zero all K-steps)
    {
        const U4 z4 = {0u, 0u, 0u, 0u};
        #pragma unroll
        for (int b = 0; b < 2; ++b) {
            if (!a0_in) *reinterpret_cast<U4*>(&As[b][arow0 * 32 + acol]) = z4;
            if (!a1_in) *reinterpret_cast<U4*>(&As[b][arow1 * 32 + acol]) = z4;
            if (!b_in)  *reinterpret_cast<U4*>(&Bs[b][brow * 32 + bcol])  = z4;
        }
    }

    // async stage of K-slab ks into LDS buffer `buf` (3 async ops per wave)
    auto issue = [&](int ks, int buf) {
        const long koff = (long)ks * 32;
        if (a0_in)
            asm volatile("global_load_async_to_lds_b128 %0, %1, off"
                         :: "v"(ldsA0[buf]),
                            "v"((unsigned long long)(size_t)(gA0 + koff))
                         : "memory");
        if (a1_in)
            asm volatile("global_load_async_to_lds_b128 %0, %1, off"
                         :: "v"(ldsA1[buf]),
                            "v"((unsigned long long)(size_t)(gA1 + koff))
                         : "memory");
        if (b_in)
            asm volatile("global_load_async_to_lds_b128 %0, %1, off"
                         :: "v"(ldsB[buf]),
                            "v"((unsigned long long)(size_t)(gB + koff))
                         : "memory");
    };

    const v8f zacc = {0.f, 0.f, 0.f, 0.f, 0.f, 0.f, 0.f, 0.f};
    v8f acc[2][2];
    acc[0][0] = zacc; acc[0][1] = zacc; acc[1][0] = zacc; acc[1][1] = zacc;

    const int m_l = lane & 15;
    const int hi  = lane >> 4;
    const int ksteps = K >> 5;

    issue(0, 0);   // prime buffer 0

    for (int s = 0; s < ksteps; ++s) {
        const int cur = s & 1;
        if (s + 1 < ksteps) {
            issue(s + 1, cur ^ 1);                 // overlap next slab fill
            // in-order async completion: <=3 outstanding => current slab landed
            asm volatile("s_wait_asynccnt 0x3" ::: "memory");
        } else {
            asm volatile("s_wait_asynccnt 0x0" ::: "memory");
        }
        __syncthreads();

        // A fragment per ISA 16-bit 16x32 layout:
        //  lanes 0-15: elems 0-7 = K0..7,  elems 8-15 = K16..23
        //  lanes16-31: elems 0-7 = K8..15, elems 8-15 = K24..31
        V16 afrag[2], bfrag[2];
        #pragma unroll
        for (int i = 0; i < 2; ++i) {
            int ar  = (wm * 32 + i * 16 + m_l) * 32;
            int klo = hi * 8;
            afrag[i].u[0] = *reinterpret_cast<const U4*>(&As[cur][ar + klo]);
            afrag[i].u[1] = *reinterpret_cast<const U4*>(&As[cur][ar + klo + 16]);
        }
        // B fragment (32x16): same K layout as A (lanes 0-15: K0..7,K16..23; lanes 16-31: K8..15,K24..31)
        #pragma unroll
        for (int j = 0; j < 2; ++j) {
            int nr = (wn * 32 + j * 16 + m_l) * 32;
            int kb = hi * 8;
            bfrag[j].u[0] = *reinterpret_cast<const U4*>(&Bs[cur][nr + kb]);
            bfrag[j].u[1] = *reinterpret_cast<const U4*>(&Bs[cur][nr + kb + 16]);
        }

        #pragma unroll
        for (int i = 0; i < 2; ++i)
            #pragma unroll
            for (int j = 0; j < 2; ++j)
                acc[i][j] = __builtin_amdgcn_wmma_f32_16x16x32_f16(
                    false, afrag[i].v, false, bfrag[j].v,
                    (short)0, acc[i][j], false, false);
        __syncthreads();   // all waves done reading `cur` before it is refilled
    }

    // epilogue -------------------------------------------------------------
    float gb = 0.f;
    if (EPI == EPI_SCORES && shift_ptr) gb = shift_ptr[0];
    float*  Cz   = C   ? C   + z * sC : nullptr;
    float*  C2z  = C2  ? C2  + z * sC : nullptr;
    half_t* C16z = C16 ? C16 + z * sC : nullptr;
    const float* Rz = resid ? resid + z * sR : nullptr;

    #pragma unroll
    for (int i = 0; i < 2; ++i)
        #pragma unroll
        for (int j = 0; j < 2; ++j)
            #pragma unroll
            for (int r = 0; r < 8; ++r) {
                int row = bm + wm * 32 + i * 16 + hi * 8 + r;
                int col = bn + wn * 32 + j * 16 + m_l;
                if (row < M && col < Nn) {
                    float v = acc[i][j][r];
                    if (bias) v += bias[col];
                    if (EPI == EPI_RELU) v = v > 0.f ? v : 0.f;
                    if (EPI == EPI_RES)  v += Rz[(long)row * ldr + col];
                    if (EPI == EPI_SCORES) {
                        v = v * scale + gb;
                        Cz[(long)row * ldc + col]  = v;
                        C2z[(long)row * ldc + col] = 1.f / (1.f + __expf(-v));
                    } else if (EPI == EPI_F16) {
                        C16z[(long)row * ldc + col] = (half_t)v;
                    } else if (EPI == EPI_RES) {
                        Cz[(long)row * ldc + col]   = v;
                        C16z[(long)row * ldc + col] = (half_t)v;
                    } else {
                        Cz[(long)row * ldc + col] = v;
                    }
                }
            }
}

// ---------------------------------------------------------------------------
// Helper kernels
// ---------------------------------------------------------------------------
__global__ void pe_kernel(half_t* __restrict__ pe16)
{
    long idx = (long)blockIdx.x * blockDim.x + threadIdx.x;
    if (idx >= (long)NN * PDD) return;
    int n = (int)(idx >> 8);
    int d = (int)(idx & 255);
    int i = d >> 1;
    float freq = __expf(-(float)(2 * i) * (logf(10000.f) / (float)PDD));
    float ang  = (float)n * freq;
    float v    = (d & 1) ? __cosf(ang) : __sinf(ang);
    pe16[idx] = (half_t)v;
}

__global__ void to_f16_kernel(const float* __restrict__ in, half_t* __restrict__ out, long n)
{
    long i = (long)blockIdx.x * blockDim.x + threadIdx.x;
    if (i < n) out[i] = (half_t)in[i];
}

// out[c*Mr + r] = in[r*Nc + c]  (row-major Mr x Nc -> row-major Nc x Mr), batched by z
__global__ void transpose_f16_kernel(const float* __restrict__ in, half_t* __restrict__ out,
                                     int Mr, int Nc)
{
    long base = (long)blockIdx.z * (long)Mr * Nc;
    long idx  = (long)blockIdx.x * blockDim.x + threadIdx.x;
    if (idx >= (long)Mr * Nc) return;
    int r = (int)(idx / Nc), c = (int)(idx % Nc);
    out[base + (long)c * Mr + r] = (half_t)in[base + idx];
}

__global__ void cat1_kernel(const half_t* __restrict__ pe16,
                            const float* __restrict__ agg1,
                            half_t* __restrict__ cat1)
{
    long idx = (long)blockIdx.x * blockDim.x + threadIdx.x;
    if (idx >= (long)ROWS * 512) return;
    long r = idx >> 9;           // t*N + n
    int  h = (int)(idx & 511);
    int  n = (int)(r & 1023);
    cat1[idx] = (h < 256) ? pe16[(long)n * 256 + h]
                          : (half_t)agg1[r * 256 + (h - 256)];
}

__global__ void cat2_kernel(const half_t* __restrict__ x16,
                            const float* __restrict__ agg2,
                            half_t* __restrict__ cat2)
{
    long idx = (long)blockIdx.x * blockDim.x + threadIdx.x;
    if (idx >= (long)ROWS * 512) return;
    long r = idx >> 9;
    int  h = (int)(idx & 511);
    cat2[idx] = (h < 256) ? x16[r * 256 + h]
                          : (half_t)agg2[r * 256 + (h - 256)];
}

// LN over rows of 256, one block (256 threads) per row, f16 output
__global__ __launch_bounds__(256)
void ln1_kernel(const float* __restrict__ in, const float* __restrict__ g,
                const float* __restrict__ b, half_t* __restrict__ out16)
{
    int row = blockIdx.x, h = threadIdx.x;
    __shared__ float red[256];
    float v = in[(long)row * 256 + h];
    red[h] = v; __syncthreads();
    for (int s = 128; s > 0; s >>= 1) { if (h < s) red[h] += red[h + s]; __syncthreads(); }
    float mu = red[0] * (1.f / 256.f); __syncthreads();
    float d  = v - mu;
    red[h] = d * d; __syncthreads();
    for (int s = 128; s > 0; s >>= 1) { if (h < s) red[h] += red[h + s]; __syncthreads(); }
    float var = red[0] * (1.f / 256.f);
    float o = d * rsqrtf(var + 1e-5f) * g[h] + b[h];
    out16[(long)row * 256 + h] = (half_t)o;
}

// LN2 (spatial) then Mamba-LN, with [t,n] -> [n,t] reorder.
// res[n*T+t] = LN2(u2[t*N+n]); xn16[n*T+t] = LN_mb(res row)
__global__ __launch_bounds__(256)
void ln2_mb_kernel(const float* __restrict__ u2,
                   const float* __restrict__ g2, const float* __restrict__ b2,
                   const float* __restrict__ gm, const float* __restrict__ bm,
                   float* __restrict__ res, half_t* __restrict__ xn16)
{
    int rin = blockIdx.x, h = threadIdx.x;        // rin = t*N + n
    int t = rin >> 10, n = rin & 1023;
    long rout = (long)n * TT + t;
    __shared__ float red[256];
    float v = u2[(long)rin * 256 + h];
    red[h] = v; __syncthreads();
    for (int s = 128; s > 0; s >>= 1) { if (h < s) red[h] += red[h + s]; __syncthreads(); }
    float mu = red[0] * (1.f / 256.f); __syncthreads();
    float d = v - mu;
    red[h] = d * d; __syncthreads();
    for (int s = 128; s > 0; s >>= 1) { if (h < s) red[h] += red[h + s]; __syncthreads(); }
    float var = red[0] * (1.f / 256.f); __syncthreads();
    float x2 = d * rsqrtf(var + 1e-5f) * g2[h] + b2[h];
    res[rout * 256 + h] = x2;
    // second LN with mamba params
    red[h] = x2; __syncthreads();
    for (int s = 128; s > 0; s >>= 1) { if (h < s) red[h] += red[h + s]; __syncthreads(); }
    float mu2 = red[0] * (1.f / 256.f); __syncthreads();
    float d2 = x2 - mu2;
    red[h] = d2 * d2; __syncthreads();
    for (int s = 128; s > 0; s >>= 1) { if (h < s) red[h] += red[h + s]; __syncthreads(); }
    float var2 = red[0] * (1.f / 256.f);
    float xn = d2 * rsqrtf(var2 + 1e-5f) * gm[h] + bm[h];
    xn16[rout * 256 + h] = (half_t)xn;
}

// silu on first half of xp -> x1 (f32 + f16)
__global__ void silu_split_kernel(const float* __restrict__ xp,
                                  float* __restrict__ x1f, half_t* __restrict__ x1h)
{
    long idx = (long)blockIdx.x * blockDim.x + threadIdx.x;
    if (idx >= (long)ROWS * EE) return;
    long r = idx / EE;
    int  e = (int)(idx % EE);
    float v = xp[r * (2 * EE) + e];
    float s = v / (1.f + __expf(-v));
    x1f[idx] = s;
    x1h[idx] = (half_t)s;
}

// delta[r][e] = softplus( sum_j dl[r][j]*dt_w[j][e] + dt_b[e] ), K=16 VALU chain
__global__ __launch_bounds__(512)
void delta_kernel(const float* __restrict__ dl, const float* __restrict__ dt_w,
                  const float* __restrict__ dt_b, float* __restrict__ delta)
{
    int r = blockIdx.x, e = threadIdx.x;
    __shared__ float sdl[DTR];
    if (e < DTR) sdl[e] = dl[(long)r * DTR + e];
    __syncthreads();
    float acc = dt_b[e];
    #pragma unroll
    for (int j = 0; j < DTR; ++j) acc += sdl[j] * dt_w[(long)j * EE + e];
    float sp = (acc > 20.f) ? acc : log1pf(__expf(acc));
    delta[(long)r * EE + e] = sp;
}

// Selective scan over T, one block per node n (512 threads = E), fused
// with D-skip connection and silu(gate) multiplication. yg = f16 for out GEMM.
__global__ __launch_bounds__(512)
void scan_kernel(const float* __restrict__ x1, const float* __restrict__ delta,
                 const float* __restrict__ Bm, const float* __restrict__ Cm,
                 const float* __restrict__ A_log, const float* __restrict__ D_vec,
                 const float* __restrict__ xp, half_t* __restrict__ yg)
{
    int n = blockIdx.x, e = threadIdx.x;
    __shared__ float sB[SDD], sC[SDD];
    float Arow[SDD], h[SDD];
    #pragma unroll
    for (int s = 0; s < SDD; ++s) {
        Arow[s] = -__expf(A_log[(long)e * SDD + s]);
        h[s] = 0.f;
    }
    float Dv = D_vec[e];
    for (int t = 0; t < TT; ++t) {
        long r = (long)n * TT + t;
        if (e < SDD)                sB[e]        = Bm[r * SDD + e];
        else if (e < 2 * SDD)       sC[e - SDD]  = Cm[r * SDD + (e - SDD)];
        __syncthreads();
        float xv = x1[r * EE + e];
        float dt = delta[r * EE + e];
        float y = 0.f;
        #pragma unroll
        for (int s = 0; s < SDD; ++s) {
            float ad = __expf(dt * Arow[s]);
            h[s] = ad * h[s] + dt * sB[s] * xv;
            y += sC[s] * h[s];
        }
        float g  = xp[r * (2 * EE) + EE + e];
        float sg = g / (1.f + __expf(-g));
        yg[r * EE + e] = (half_t)((y + xv * Dv) * sg);
        __syncthreads();
    }
}

// outs[n*T+t][h] (f32) -> d_out outs region [t*N+n][h]
__global__ void reorder_out_kernel(const float* __restrict__ outs, float* __restrict__ dst)
{
    long idx = (long)blockIdx.x * blockDim.x + threadIdx.x;
    if (idx >= (long)ROWS * HH) return;
    long r = idx >> 8;                 // n*T + t
    int  h = (int)(idx & 255);
    int  n = (int)(r >> 5), t = (int)(r & 31);
    dst[((long)t * NN + n) * HH + h] = outs[idx];
}

// ---------------------------------------------------------------------------
// Host side
// ---------------------------------------------------------------------------
static inline unsigned cdivu(long a, long b) { return (unsigned)((a + b - 1) / b); }

extern "C" void kernel_launch(void* const* d_in, const int* in_sizes, int n_in,
                              void* d_out, int out_size, void* d_ws, size_t ws_size,
                              hipStream_t stream)
{
    (void)in_sizes; (void)n_in; (void)out_size; (void)ws_size;

    const float* adj     = (const float*)d_in[0];
    const float* msg1_w  = (const float*)d_in[1];  const float* msg1_b = (const float*)d_in[2];
    const float* upd1_w  = (const float*)d_in[3];  const float* upd1_b = (const float*)d_in[4];
    const float* ln1_g   = (const float*)d_in[5];  const float* ln1_b  = (const float*)d_in[6];
    const float* msg2_w  = (const float*)d_in[7];  const float* msg2_b = (const float*)d_in[8];
    const float* upd2_w  = (const float*)d_in[9];  const float* upd2_b = (const float*)d_in[10];
    const float* ln2_g   = (const float*)d_in[11]; const float* ln2_b  = (const float*)d_in[12];
    const float* mb_ln_g = (const float*)d_in[13]; const float* mb_ln_b= (const float*)d_in[14];
    const float* in_w    = (const float*)d_in[15]; const float* in_b   = (const float*)d_in[16];
    const float* delta_w = (const float*)d_in[17]; const float* delta_b= (const float*)d_in[18];
    const float* dt_w    = (const float*)d_in[19]; const float* dt_b   = (const float*)d_in[20];
    const float* Bp_w    = (const float*)d_in[21]; const float* Bp_b   = (const float*)d_in[22];
    const float* Cp_w    = (const float*)d_in[23]; const float* Cp_b   = (const float*)d_in[24];
    const float* A_log   = (const float*)d_in[25]; const float* D_vec  = (const float*)d_in[26];
    const float* out_w   = (const float*)d_in[27]; const float* out_b  = (const float*)d_in[28];
    const float* q_w     = (const float*)d_in[29]; const float* k_w    = (const float*)d_in[30];
    const float* gate_bias = (const float*)d_in[31];

    float* out = (float*)d_out;
    float* out_outs   = out;                                  // [T,N,H]
    float* out_probs  = out + (long)TT * NN * HH;             // [T,N,N]
    float* out_logits = out_probs + (long)TT * NN * NN;       // [T,N,N]

    // ---- workspace bump allocator -------------------------------------
    char* ws = (char*)d_ws;
    size_t off = 0;
    auto alloc = [&](size_t bytes) -> void* {
        off = (off + 255) & ~(size_t)255;
        void* p = ws + off;
        off += bytes;
        return p;
    };
    half_t* pe16   = (half_t*)alloc((size_t)NN * PDD * 2);
    half_t* adj16  = (half_t*)alloc((size_t)TT * NN * NN * 2);
    half_t* msg1T  = (half_t*)alloc((size_t)PDD * HH * 2);
    half_t* upd1T  = (half_t*)alloc((size_t)512 * HH * 2);
    half_t* msg2T  = (half_t*)alloc((size_t)HH * HH * 2);
    half_t* upd2T  = (half_t*)alloc((size_t)512 * HH * 2);
    half_t* inT    = (half_t*)alloc((size_t)HH * 2 * EE * 2);
    half_t* deltaT = (half_t*)alloc((size_t)EE * DTR * 2);
    half_t* BpT    = (half_t*)alloc((size_t)EE * SDD * 2);
    half_t* CpT    = (half_t*)alloc((size_t)EE * SDD * 2);
    half_t* outT   = (half_t*)alloc((size_t)EE * HH * 2);
    half_t* qT     = (half_t*)alloc((size_t)HH * HH * 2);
    half_t* kT     = (half_t*)alloc((size_t)HH * HH * 2);
    float*  m1f    = (float*) alloc((size_t)NN * HH * 4);
    half_t* m1T    = (half_t*)alloc((size_t)NN * HH * 2);
    float*  agg1f  = (float*) alloc((size_t)ROWS * HH * 4);
    half_t* cat1h  = (half_t*)alloc((size_t)ROWS * 512 * 2);
    float*  u1f    = (float*) alloc((size_t)ROWS * HH * 4);
    half_t* x16    = (half_t*)alloc((size_t)ROWS * HH * 2);
    float*  m2f    = (float*) alloc((size_t)ROWS * HH * 4);
    half_t* m2T    = (half_t*)alloc((size_t)TT * HH * NN * 2);
    float*  agg2f  = (float*) alloc((size_t)ROWS * HH * 4);
    half_t* cat2h  = (half_t*)alloc((size_t)ROWS * 512 * 2);
    float*  u2f    = (float*) alloc((size_t)ROWS * HH * 4);
    float*  resf   = (float*) alloc((size_t)ROWS * HH * 4);
    half_t* xn16   = (half_t*)alloc((size_t)ROWS * HH * 2);
    float*  xpf    = (float*) alloc((size_t)ROWS * 2 * EE * 4);
    float*  x1f    = (float*) alloc((size_t)ROWS * EE * 4);
    half_t* x1h    = (half_t*)alloc((size_t)ROWS * EE * 2);
    float*  dlf    = (float*) alloc((size_t)ROWS * DTR * 4);
    float*  deltaf = (float*) alloc((size_t)ROWS * EE * 4);
    float*  Bmf    = (float*) alloc((size_t)ROWS * SDD * 4);
    float*  Cmf    = (float*) alloc((size_t)ROWS * SDD * 4);
    half_t* ygh    = (half_t*)alloc((size_t)ROWS * EE * 2);
    float*  outsf  = (float*) alloc((size_t)ROWS * HH * 4);
    half_t* outs16 = (half_t*)alloc((size_t)ROWS * HH * 2);
    half_t* q16    = (half_t*)alloc((size_t)ROWS * HH * 2);
    half_t* k16    = (half_t*)alloc((size_t)ROWS * HH * 2);

    const dim3 TB(256);

    // ---- precompute: PE, f16 adj, transposed f16 weights ---------------
    pe_kernel<<<cdivu((long)NN * PDD, 256), TB, 0, stream>>>(pe16);
    to_f16_kernel<<<cdivu((long)TT * NN * NN, 256), TB, 0, stream>>>(adj, adj16, (long)TT * NN * NN);
    transpose_f16_kernel<<<dim3(cdivu((long)PDD * HH, 256), 1, 1), TB, 0, stream>>>(msg1_w, msg1T, PDD, HH);
    transpose_f16_kernel<<<dim3(cdivu((long)512 * HH, 256), 1, 1), TB, 0, stream>>>(upd1_w, upd1T, 512, HH);
    transpose_f16_kernel<<<dim3(cdivu((long)HH * HH, 256), 1, 1), TB, 0, stream>>>(msg2_w, msg2T, HH, HH);
    transpose_f16_kernel<<<dim3(cdivu((long)512 * HH, 256), 1, 1), TB, 0, stream>>>(upd2_w, upd2T, 512, HH);
    transpose_f16_kernel<<<dim3(cdivu((long)HH * 2 * EE, 256), 1, 1), TB, 0, stream>>>(in_w, inT, HH, 2 * EE);
    transpose_f16_kernel<<<dim3(cdivu((long)EE * DTR, 256), 1, 1), TB, 0, stream>>>(delta_w, deltaT, EE, DTR);
    transpose_f16_kernel<<<dim3(cdivu((long)EE * SDD, 256), 1, 1), TB, 0, stream>>>(Bp_w, BpT, EE, SDD);
    transpose_f16_kernel<<<dim3(cdivu((long)EE * SDD, 256), 1, 1), TB, 0, stream>>>(Cp_w, CpT, EE, SDD);
    transpose_f16_kernel<<<dim3(cdivu((long)EE * HH, 256), 1, 1), TB, 0, stream>>>(out_w, outT, EE, HH);
    transpose_f16_kernel<<<dim3(cdivu((long)HH * HH, 256), 1, 1), TB, 0, stream>>>(q_w, qT, HH, HH);
    transpose_f16_kernel<<<dim3(cdivu((long)HH * HH, 256), 1, 1), TB, 0, stream>>>(k_w, kT, HH, HH);

    // ---- spatial layer 1 ----------------------------------------------
    // m1 = pe @ msg1_w + b
    wmma_gemm_kernel<EPI_PLAIN><<<dim3(cdivu(HH,64), cdivu(NN,128), 1), TB, 0, stream>>>(
        pe16, PDD, 0, msg1T, PDD, 0, m1f, HH, 0, nullptr, msg1_b,
        nullptr, 0, 0, nullptr, 1.f, nullptr, NN, HH, PDD);
    transpose_f16_kernel<<<dim3(cdivu((long)NN * HH, 256), 1, 1), TB, 0, stream>>>(m1f, m1T, NN, HH);
    // agg1[t] = adj16[t] @ m1
    wmma_gemm_kernel<EPI_PLAIN><<<dim3(cdivu(HH,64), cdivu(NN,128), TT), TB, 0, stream>>>(
        adj16, NN, (long)NN * NN, m1T, NN, 0, agg1f, HH, (long)NN * HH, nullptr, nullptr,
        nullptr, 0, 0, nullptr, 1.f, nullptr, NN, HH, NN);
    cat1_kernel<<<cdivu((long)ROWS * 512, 256), TB, 0, stream>>>(pe16, agg1f, cat1h);
    // u1 = relu(cat1 @ upd1_w + b)
    wmma_gemm_kernel<EPI_RELU><<<dim3(cdivu(HH,64), cdivu(ROWS,128), 1), TB, 0, stream>>>(
        cat1h, 512, 0, upd1T, 512, 0, u1f, HH, 0, nullptr, upd1_b,
        nullptr, 0, 0, nullptr, 1.f, nullptr, ROWS, HH, 512);
    ln1_kernel<<<ROWS, TB, 0, stream>>>(u1f, ln1_g, ln1_b, x16);

    // ---- spatial layer 2 ----------------------------------------------
    wmma_gemm_kernel<EPI_PLAIN><<<dim3(cdivu(HH,64), cdivu(ROWS,128), 1), TB, 0, stream>>>(
        x16, HH, 0, msg2T, HH, 0, m2f, HH, 0, nullptr, msg2_b,
        nullptr, 0, 0, nullptr, 1.f, nullptr, ROWS, HH, HH);
    transpose_f16_kernel<<<dim3(cdivu((long)NN * HH, 256), 1, TT), TB, 0, stream>>>(m2f, m2T, NN, HH);
    wmma_gemm_kernel<EPI_PLAIN><<<dim3(cdivu(HH,64), cdivu(NN,128), TT), TB, 0, stream>>>(
        adj16, NN, (long)NN * NN, m2T, NN, (long)HH * NN, agg2f, HH, (long)NN * HH, nullptr, nullptr,
        nullptr, 0, 0, nullptr, 1.f, nullptr, NN, HH, NN);
    cat2_kernel<<<cdivu((long)ROWS * 512, 256), TB, 0, stream>>>(x16, agg2f, cat2h);
    wmma_gemm_kernel<EPI_RELU><<<dim3(cdivu(HH,64), cdivu(ROWS,128), 1), TB, 0, stream>>>(
        cat2h, 512, 0, upd2T, 512, 0, u2f, HH, 0, nullptr, upd2_b,
        nullptr, 0, 0, nullptr, 1.f, nullptr, ROWS, HH, 512);
    ln2_mb_kernel<<<ROWS, TB, 0, stream>>>(u2f, ln2_g, ln2_b, mb_ln_g, mb_ln_b, resf, xn16);

    // ---- mamba ---------------------------------------------------------
    wmma_gemm_kernel<EPI_PLAIN><<<dim3(cdivu(2*EE,64), cdivu(ROWS,128), 1), TB, 0, stream>>>(
        xn16, HH, 0, inT, HH, 0, xpf, 2 * EE, 0, nullptr, in_b,
        nullptr, 0, 0, nullptr, 1.f, nullptr, ROWS, 2 * EE, HH);
    silu_split_kernel<<<cdivu((long)ROWS * EE, 256), TB, 0, stream>>>(xpf, x1f, x1h);
    wmma_gemm_kernel<EPI_PLAIN><<<dim3(cdivu(DTR,64), cdivu(ROWS,128), 1), TB, 0, stream>>>(
        x1h, EE, 0, deltaT, EE, 0, dlf, DTR, 0, nullptr, delta_b,
        nullptr, 0, 0, nullptr, 1.f, nullptr, ROWS, DTR, EE);
    delta_kernel<<<ROWS, dim3(512), 0, stream>>>(dlf, dt_w, dt_b, deltaf);
    wmma_gemm_kernel<EPI_PLAIN><<<dim3(cdivu(SDD,64), cdivu(ROWS,128), 1), TB, 0, stream>>>(
        x1h, EE, 0, BpT, EE, 0, Bmf, SDD, 0, nullptr, Bp_b,
        nullptr, 0, 0, nullptr, 1.f, nullptr, ROWS, SDD, EE);
    wmma_gemm_kernel<EPI_PLAIN><<<dim3(cdivu(SDD,64), cdivu(ROWS,128), 1), TB, 0, stream>>>(
        x1h, EE, 0, CpT, EE, 0, Cmf, SDD, 0, nullptr, Cp_b,
        nullptr, 0, 0, nullptr, 1.f, nullptr, ROWS, SDD, EE);
    scan_kernel<<<NN, dim3(512), 0, stream>>>(x1f, deltaf, Bmf, Cmf, A_log, D_vec, xpf, ygh);
    // out proj + residual, dual f32/f16 store
    wmma_gemm_kernel<EPI_RES><<<dim3(cdivu(HH,64), cdivu(ROWS,128), 1), TB, 0, stream>>>(
        ygh, EE, 0, outT, EE, 0, outsf, HH, 0, outs16, out_b,
        resf, HH, 0, nullptr, 1.f, nullptr, ROWS, HH, EE);
    reorder_out_kernel<<<cdivu((long)ROWS * HH, 256), TB, 0, stream>>>(outsf, out_outs);

    // ---- late gate head -------------------------------------------------
    wmma_gemm_kernel<EPI_F16><<<dim3(cdivu(HH,64), cdivu(ROWS,128), 1), TB, 0, stream>>>(
        outs16, HH, 0, qT, HH, 0, nullptr, HH, 0, q16, nullptr,
        nullptr, 0, 0, nullptr, 1.f, nullptr, ROWS, HH, HH);
    wmma_gemm_kernel<EPI_F16><<<dim3(cdivu(HH,64), cdivu(ROWS,128), 1), TB, 0, stream>>>(
        outs16, HH, 0, kT, HH, 0, nullptr, HH, 0, k16, nullptr,
        nullptr, 0, 0, nullptr, 1.f, nullptr, ROWS, HH, HH);
    // scores[t] = q[t] @ k[t]^T * (1/16) + gate_bias ; logits + sigmoid probs
    const float scale = 1.f / 16.f;  // 1/(sqrt(256)*TAU)
    wmma_gemm_kernel<EPI_SCORES><<<dim3(cdivu(NN,64), cdivu(NN,128), TT), TB, 0, stream>>>(
        q16, (long)TT * HH, HH, k16, (long)TT * HH, HH,
        out_logits, NN, (long)NN * NN, nullptr, nullptr,
        nullptr, 0, 0, gate_bias, scale, out_probs, NN, NN, HH);
}
